// KNRM_86406152061728
// MI455X (gfx1250) — matrix operand order repaired and checked by reading, not compiled, not run on hardware
//
#include <hip/hip_runtime.h>
#include <hip/hip_bf16.h>
#include <math.h>

typedef __attribute__((ext_vector_type(16))) _Float16 v16h;
typedef __attribute__((ext_vector_type(8)))  _Float16 v8h;
typedef __attribute__((ext_vector_type(8)))  float    v8f;

#define KNUM 21
#define BQ   32
#define BD   128
#define DIMD 128

// mus per KNRM._get_kernels_layers (kernel_num=21): step=0.1
__constant__ float c_mus[KNUM] = {
    -0.95f, -0.85f, -0.75f, -0.65f, -0.55f, -0.45f, -0.35f, -0.25f, -0.15f, -0.05f,
     0.05f,  0.15f,  0.25f,  0.35f,  0.45f,  0.55f,  0.65f,  0.75f,  0.85f,  0.95f,
     1.00f };
// -1/(2*sigma^2): sigma=0.1 for first 20, 0.001 for exact-match kernel
__constant__ float c_coef[KNUM] = {
    -50.f, -50.f, -50.f, -50.f, -50.f, -50.f, -50.f, -50.f, -50.f, -50.f,
    -50.f, -50.f, -50.f, -50.f, -50.f, -50.f, -50.f, -50.f, -50.f, -50.f,
    -500000.f };

__global__ __launch_bounds__(256)
void knrm_fused_kernel(const float* __restrict__ emb,
                       const int* __restrict__ q1, const int* __restrict__ d1,
                       const int* __restrict__ q2, const int* __restrict__ d2,
                       const float* __restrict__ W1, const float* __restrict__ b1,
                       const float* __restrict__ W2, const float* __restrict__ b2,
                       const float* __restrict__ W3, const float* __restrict__ b3,
                       float* __restrict__ out)
{
    __shared__ _Float16 Aq[BQ][DIMD];      //  8 KB  normalized query rows (f16)
    __shared__ _Float16 Bd[BD][DIMD];      // 32 KB  normalized doc rows (f16)
    __shared__ float    Msh[BQ][BD];       // 16 KB  similarity tile
    __shared__ float    ksum[BQ][KNUM];    //  per-query-row kernel sums over LD
    __shared__ float    koutS[KNUM];
    __shared__ float    logitS[2];

    const int b     = blockIdx.x;
    const int tid   = threadIdx.x;
    const int wv    = tid >> 5;       // wave 0..7
    const int lane  = tid & 31;
    const int lhalf = lane >> 4;      // 0/1 half-wave
    const int lmod  = lane & 15;

    const int* qsel[2] = { q1, q2 };
    const int* dsel[2] = { d1, d2 };

    for (int p = 0; p < 2; ++p) {
        // ---- zero per-pair kernel accumulators ----
        for (int i = tid; i < BQ * KNUM; i += 256)
            (&ksum[0][0])[i] = 0.0f;

        // ---- gather + L2-normalize: one wave per embedding row ----
        const int* qidx = qsel[p] + (long)b * BQ;
        for (int r = wv; r < BQ; r += 8) {
            long tok = (long)qidx[r];
            float4 v = reinterpret_cast<const float4*>(emb + tok * (long)DIMD)[lane];
            float ss = v.x*v.x + v.y*v.y + v.z*v.z + v.w*v.w;
            #pragma unroll
            for (int off = 16; off > 0; off >>= 1) ss += __shfl_xor(ss, off);
            float inv = 1.0f / fmaxf(sqrtf(ss), 1e-12f);
            Aq[r][lane*4+0] = (_Float16)(v.x * inv);
            Aq[r][lane*4+1] = (_Float16)(v.y * inv);
            Aq[r][lane*4+2] = (_Float16)(v.z * inv);
            Aq[r][lane*4+3] = (_Float16)(v.w * inv);
        }
        const int* didx = dsel[p] + (long)b * BD;
        for (int r = wv; r < BD; r += 8) {
            long tok = (long)didx[r];
            float4 v = reinterpret_cast<const float4*>(emb + tok * (long)DIMD)[lane];
            float ss = v.x*v.x + v.y*v.y + v.z*v.z + v.w*v.w;
            #pragma unroll
            for (int off = 16; off > 0; off >>= 1) ss += __shfl_xor(ss, off);
            float inv = 1.0f / fmaxf(sqrtf(ss), 1e-12f);
            Bd[r][lane*4+0] = (_Float16)(v.x * inv);
            Bd[r][lane*4+1] = (_Float16)(v.y * inv);
            Bd[r][lane*4+2] = (_Float16)(v.z * inv);
            Bd[r][lane*4+3] = (_Float16)(v.w * inv);
        }
        __syncthreads();

        // ---- WMMA GEMM: M[32,128] = Aq(32x128) * Bd(128x128)^T ----
        // wave wv owns N-tile [wv*16, wv*16+16); M-tiles 0 (rows 0-15) and 1 (rows 16-31)
        {
            const int n0 = wv * 16;
            v8f acc0 = {};
            v8f acc1 = {};
            #pragma unroll
            for (int kk = 0; kk < 4; ++kk) {
                const int kb = kk * 32;
                // B fragment (K x N = 32x16): lane = n, two 8-half runs at K = lhalf*8 and 16 + lhalf*8
                v8h blo = *reinterpret_cast<const v8h*>(&Bd[n0 + lmod][kb      + lhalf*8]);
                v8h bhi = *reinterpret_cast<const v8h*>(&Bd[n0 + lmod][kb + 16 + lhalf*8]);
                v16h bf = __builtin_shufflevector(blo, bhi, 0,1,2,3,4,5,6,7,8,9,10,11,12,13,14,15);
                // A fragments (M x K = 16x32): lane = m, two contiguous 8-half runs
                v8h a0lo = *reinterpret_cast<const v8h*>(&Aq[lmod     ][kb      + lhalf*8]);
                v8h a0hi = *reinterpret_cast<const v8h*>(&Aq[lmod     ][kb + 16 + lhalf*8]);
                v8h a1lo = *reinterpret_cast<const v8h*>(&Aq[16 + lmod][kb      + lhalf*8]);
                v8h a1hi = *reinterpret_cast<const v8h*>(&Aq[16 + lmod][kb + 16 + lhalf*8]);
                v16h a0 = __builtin_shufflevector(a0lo, a0hi, 0,1,2,3,4,5,6,7,8,9,10,11,12,13,14,15);
                v16h a1 = __builtin_shufflevector(a1lo, a1hi, 0,1,2,3,4,5,6,7,8,9,10,11,12,13,14,15);
                acc0 = __builtin_amdgcn_wmma_f32_16x16x32_f16(false, a0, false, bf,
                                                              (short)0, acc0, false, false);
                acc1 = __builtin_amdgcn_wmma_f32_16x16x32_f16(false, a1, false, bf,
                                                              (short)0, acc1, false, false);
            }
            // C layout: lanes 0-15 -> M = j, lanes 16-31 -> M = 8+j; N = lmod
            #pragma unroll
            for (int j = 0; j < 8; ++j) {
                Msh[     lhalf*8 + j][n0 + lmod] = acc0[j];
                Msh[16 + lhalf*8 + j][n0 + lmod] = acc1[j];
            }
        }
        __syncthreads();

        // ---- 21 Gaussian kernel features, summed over LD per query row ----
        {
            const int l  = tid >> 3;   // query row 0..31
            const int c8 = tid & 7;    // chunk of 16 doc positions
            float part[KNUM];
            #pragma unroll
            for (int k = 0; k < KNUM; ++k) part[k] = 0.0f;
            #pragma unroll 4
            for (int rr = 0; rr < 16; ++rr) {
                float x = Msh[l][c8*16 + rr];
                #pragma unroll
                for (int k = 0; k < KNUM; ++k) {
                    float dlt = x - c_mus[k];
                    part[k] += __expf(dlt * dlt * c_coef[k]);
                }
            }
            #pragma unroll
            for (int k = 0; k < KNUM; ++k)
                atomicAdd(&ksum[l][k], part[k]);
        }
        __syncthreads();

        // ---- log1p + sum over LQ ----
        if (tid < KNUM) {
            float s = 0.0f;
            for (int l = 0; l < BQ; ++l)
                s += __logf(1.0f + ksum[l][tid]);
            koutS[tid] = s;
        }
        __syncthreads();

        // ---- tiny MLP 21 -> 10 -> 5 -> 1 ----
        if (tid == 0) {
            float h1[10];
            #pragma unroll
            for (int j = 0; j < 10; ++j) {
                float a = b1[j];
                for (int k = 0; k < KNUM; ++k) a += koutS[k] * W1[k*10 + j];
                h1[j] = fmaxf(a, 0.0f);
            }
            float h2[5];
            #pragma unroll
            for (int j = 0; j < 5; ++j) {
                float a = b2[j];
                for (int k = 0; k < 10; ++k) a += h1[k] * W2[k*5 + j];
                h2[j] = fmaxf(a, 0.0f);
            }
            float lg = b3[0];
            #pragma unroll
            for (int k = 0; k < 5; ++k) lg += h2[k] * W3[k];
            logitS[p] = lg;
        }
        __syncthreads();
    }

    if (tid == 0) {
        float z = logitS[0] - logitS[1];
        out[b] = 1.0f / (1.0f + __expf(-z));
    }
}

extern "C" void kernel_launch(void* const* d_in, const int* in_sizes, int n_in,
                              void* d_out, int out_size, void* d_ws, size_t ws_size,
                              hipStream_t stream) {
    (void)in_sizes; (void)n_in; (void)out_size; (void)d_ws; (void)ws_size;
    const float* emb = (const float*)d_in[0];
    const int*   q1  = (const int*)d_in[1];
    const int*   d1  = (const int*)d_in[2];
    const int*   q2  = (const int*)d_in[3];
    const int*   d2  = (const int*)d_in[4];
    const float* W1  = (const float*)d_in[5];
    const float* b1  = (const float*)d_in[6];
    const float* W2  = (const float*)d_in[7];
    const float* b2  = (const float*)d_in[8];
    const float* W3  = (const float*)d_in[9];
    const float* b3  = (const float*)d_in[10];
    float* out = (float*)d_out;

    hipLaunchKernelGGL(knrm_fused_kernel, dim3(4096), dim3(256), 0, stream,
                       emb, q1, d1, q2, d2, W1, b1, W2, b2, W3, b3, out);
}
